// GraphConvLSTMCell_47802986005059
// MI455X (gfx1250) — matrix operand order repaired and checked by reading, not compiled, not run on hardware
//
#include <hip/hip_runtime.h>
#include <hip/hip_bf16.h>

typedef __attribute__((ext_vector_type(16))) _Float16 v16h;
typedef __attribute__((ext_vector_type(4)))  _Float16 v4h;
typedef __attribute__((ext_vector_type(8)))  float    v8f;

#define NB_T 32000
#define B_T  32
#define N_T  1000
#define H_T  256
#define D_T  128
#define E_T  512000
#define WG_STRIDE 264              // 256 + 8 halfs pad -> 132 dwords/row, conflict-free b128 reads
#define OUT_OFF 8192000            // B*N*H

static __device__ __forceinline__ void atomicAddF(float* p, float v) {
  __hip_atomic_fetch_add(p, v, __ATOMIC_RELAXED, __HIP_MEMORY_SCOPE_AGENT);
}

__global__ void k_zero(float* agg, float* dego, float* degi) {
  int i = blockIdx.x * blockDim.x + threadIdx.x;
  if (i < NB_T * H_T) agg[i] = 0.0f;
  if (i < NB_T) { dego[i] = 0.0f; degi[i] = 0.0f; }
}

__global__ void k_degree(const int* __restrict__ src, const int* __restrict__ dst,
                         float* dego, float* degi) {
  int e = blockIdx.x * blockDim.x + threadIdx.x;
  if (e < E_T) {
    atomicAddF(&dego[src[e]], 1.0f);
    atomicAddF(&degi[dst[e]], 1.0f);
  }
}

__global__ void k_rsqrt(float* dego, float* degi) {
  int i = blockIdx.x * blockDim.x + threadIdx.x;
  if (i < NB_T) {
    dego[i] = rsqrtf(fmaxf(dego[i], 1.0f));
    degi[i] = rsqrtf(fmaxf(degi[i], 1.0f));
  }
}

// xg[g][b][h] = x[b,:] @ W_g[:,h] + b_g[h]   (tiny: 4*32*256 dots of length 128)
__global__ void k_xproj(const float* __restrict__ x,
                        const float* __restrict__ Wi, const float* __restrict__ bi,
                        const float* __restrict__ Wf, const float* __restrict__ bf,
                        const float* __restrict__ Wo, const float* __restrict__ bo,
                        const float* __restrict__ Wc, const float* __restrict__ bc,
                        float* __restrict__ xg) {
  int t = blockIdx.x * blockDim.x + threadIdx.x;
  if (t >= 4 * B_T * H_T) return;
  int g = t >> 13, b = (t >> 8) & 31, h = t & 255;
  const float* W; const float* bias;
  switch (g) {
    case 0:  W = Wi; bias = bi; break;
    case 1:  W = Wf; bias = bf; break;
    case 2:  W = Wo; bias = bo; break;
    default: W = Wc; bias = bc; break;
  }
  float acc = bias[h];
  for (int k = 0; k < D_T; ++k) acc += x[b * D_T + k] * W[k * H_T + h];
  xg[t] = acc;
}

// one thread per (edge, 4-feature chunk): agg[dst] += h_prev[src] * deg_out^-1/2[src]
__global__ void k_scatter(const int* __restrict__ src, const int* __restrict__ dst,
                          const float* __restrict__ h_prev, const float* __restrict__ dego,
                          float* __restrict__ agg) {
  int t = blockIdx.x * blockDim.x + threadIdx.x;   // E*64 threads, exact grid
  int e = t >> 6;
  int c = (t & 63) << 2;
  int s = src[e], d = dst[e];
  float sc = dego[s];
  const float4 hv = *(const float4*)(h_prev + (size_t)s * H_T + c);
  float* p = agg + (size_t)d * H_T + c;
  atomicAddF(p + 0, hv.x * sc);
  atomicAddF(p + 1, hv.y * sc);
  atomicAddF(p + 2, hv.z * sc);
  atomicAddF(p + 3, hv.w * sc);
}

// aggH = f16( agg * deg_in^-1/2[row] )
__global__ void k_normcvt(const float* __restrict__ agg, const float* __restrict__ degi,
                          _Float16* __restrict__ aggH) {
  int i = blockIdx.x * blockDim.x + threadIdx.x;
  if (i < NB_T * H_T) aggH[i] = (_Float16)(agg[i] * degi[i >> 8]);
}

// Fused: h_conv = aggH @ Wg + bg  (WMMA f16->f32), then LSTM gates -> h_t, c_t
__global__ void __launch_bounds__(256) k_gemm_lstm(
    const _Float16* __restrict__ aggH, const float* __restrict__ Wg,
    const float* __restrict__ bg, const float* __restrict__ xg,
    const float* __restrict__ c_prev, float* __restrict__ out) {
  extern __shared__ char smem_raw[];
  _Float16* wg_s = (_Float16*)smem_raw;                          // 256 rows x 264 halfs (padded)
  float* bg_s = (float*)(smem_raw + 2 * 256 * WG_STRIDE);        // 256 floats

  const int tid = threadIdx.x;
  // stage Wg (f32 row-major K x N) -> f16 LDS with padded rows
  for (int i = tid; i < (H_T * H_T) / 4; i += 256) {
    int k = i >> 6;                 // i*4 / 256
    int n = (i & 63) << 2;
    float4 w = *(const float4*)(Wg + (size_t)k * H_T + n);
    v4h h4;
    h4[0] = (_Float16)w.x; h4[1] = (_Float16)w.y;
    h4[2] = (_Float16)w.z; h4[3] = (_Float16)w.w;
    *(v4h*)(wg_s + k * WG_STRIDE + n) = h4;
  }
  bg_s[tid & 255] = bg[tid & 255];
  __syncthreads();

  const int lane = tid & 31;
  const int wave = tid >> 5;
  const int mbase = blockIdx.x * 128 + wave * 16;       // 250 blocks * 8 waves * 16 rows = 32000
  const int m = mbase + (lane & 15);                    // A-frag: lane holds row m = lane%16
  const int aoff = (lane >> 4) * 8;                     // K sub-chunk select per lane half

  v8f acc[16];
  const v8f vzero = {0.f, 0.f, 0.f, 0.f, 0.f, 0.f, 0.f, 0.f};
  #pragma unroll
  for (int i = 0; i < 16; ++i) acc[i] = vzero;

  const _Float16* arow = aggH + (size_t)m * H_T;
  for (int ks = 0; ks < 8; ++ks) {                      // K = 256 = 8 * 32
    const int k0 = ks * 32;
    union { v16h v; uint4 u[2]; } A;
    A.u[0] = *(const uint4*)(arow + k0 + aoff);         // K = k0+off .. +7
    A.u[1] = *(const uint4*)(arow + k0 + aoff + 16);    // K = k0+off+16 .. +23
    const _Float16* brow = wg_s + (k0 + aoff) * WG_STRIDE + (lane & 15);  // B-frag: lane holds col n=lane%16, K=k0+off.. (+16)
    #pragma unroll
    for (int nt = 0; nt < 16; ++nt) {
      union { v16h v; uint4 u[2]; } Bf;
      #pragma unroll
      for (int s = 0; s < 16; ++s)
        Bf.v[s] = brow[((s < 8) ? s : s + 8) * WG_STRIDE + nt * 16];
      acc[nt] = __builtin_amdgcn_wmma_f32_16x16x32_f16(
          false, A.v, false, Bf.v, (short)0, acc[nt], false, false);
    }
  }

  // C-frag layout: element (row = mbase + 8*(lane/16) + r, col = nt*16 + lane%16) in acc[nt][r]
  const int rbase = mbase + ((lane >> 4) << 3);
  #pragma unroll
  for (int nt = 0; nt < 16; ++nt) {
    const int col = nt * 16 + (lane & 15);
    const float bgv = bg_s[col];
    #pragma unroll
    for (int r = 0; r < 8; ++r) {
      const int node = rbase + r;
      const int b = node / N_T;
      const float hc = acc[nt][r] + bgv;
      const float xiv = xg[b * H_T + col];
      const float xfv = xg[8192 + b * H_T + col];
      const float xov = xg[16384 + b * H_T + col];
      const float xcv = xg[24576 + b * H_T + col];
      const float it = 1.0f / (1.0f + __expf(-(xiv + hc)));
      const float ft = 1.0f / (1.0f + __expf(-(xfv + hc)));
      const float ot = 1.0f / (1.0f + __expf(-(xov + hc)));
      const float ctil = tanhf(xcv + hc);
      const float ct = ft * c_prev[(size_t)node * H_T + col] + it * ctil;
      const float ht = ot * tanhf(ct);
      out[(size_t)node * H_T + col] = ht;
      out[(size_t)OUT_OFF + (size_t)node * H_T + col] = ct;
    }
  }
}

extern "C" void kernel_launch(void* const* d_in, const int* in_sizes, int n_in,
                              void* d_out, int out_size, void* d_ws, size_t ws_size,
                              hipStream_t stream) {
  const float* x      = (const float*)d_in[0];
  const float* h_prev = (const float*)d_in[1];
  const float* c_prev = (const float*)d_in[2];
  const int*   src    = (const int*)d_in[3];
  const int*   dst    = (const int*)d_in[4];
  const float* Wi = (const float*)d_in[5];  const float* bi = (const float*)d_in[6];
  const float* Wf = (const float*)d_in[7];  const float* bf = (const float*)d_in[8];
  const float* Wo = (const float*)d_in[9];  const float* bo = (const float*)d_in[10];
  const float* Wc = (const float*)d_in[11]; const float* bc = (const float*)d_in[12];
  const float* Wg = (const float*)d_in[13]; const float* bg = (const float*)d_in[14];

  char* ws = (char*)d_ws;
  float*    agg  = (float*)ws;                          // 32000*256*4 = 32,768,000 B
  _Float16* aggH = (_Float16*)(ws + 32768000);          // 16,384,000 B
  float*    dego = (float*)(ws + 49152000);             // 128,000 B
  float*    degi = (float*)(ws + 49280000);             // 128,000 B
  float*    xg   = (float*)(ws + 49408000);             // 4*32*256*4 = 131,072 B
  float*    out  = (float*)d_out;

  k_zero   <<<(NB_T * H_T) / 256, 256, 0, stream>>>(agg, dego, degi);
  k_degree <<<(E_T + 255) / 256, 256, 0, stream>>>(src, dst, dego, degi);
  k_rsqrt  <<<(NB_T + 255) / 256, 256, 0, stream>>>(dego, degi);
  k_xproj  <<<(4 * B_T * H_T) / 256, 256, 0, stream>>>(x, Wi, bi, Wf, bf, Wo, bo, Wc, bc, xg);
  k_scatter<<<(E_T * 64) / 256, 256, 0, stream>>>(src, dst, h_prev, dego, agg);
  k_normcvt<<<(NB_T * H_T) / 256, 256, 0, stream>>>(agg, degi, aggH);

  const size_t smem_bytes = 2 * 256 * WG_STRIDE + 1024; // 135,168 + 1,024 = 136,192 B (< 320 KB/WGP)
  k_gemm_lstm<<<NB_T / 128, 256, smem_bytes, stream>>>(aggH, Wg, bg, xg, c_prev, out);
}
